// QuantLinear_28484223107806
// MI455X (gfx1250) — matrix operand order repaired and checked by reading, not compiled, not run on hardware
//
#include <hip/hip_runtime.h>
#include <hip/hip_bf16.h>

// ---------------------------------------------------------------------------
// QuantLinear: out[M,N] = A[M,K] @ (scale[N] * (Q[N,K] - 128))^T + bias[N]
// M = B*S = 512, K = 4096, N = 11008.
// fp16 WMMA GEMM, on-the-fly dequant (q-128 exact in f16, scale in epilogue).
// Pipelined: global->reg stage of tile k+1 overlaps WMMA of tile k; ping-pong
// LDS, one barrier per K-step. Wave tile 64Mx32N: 4 resident A fragments,
// each B fragment feeds 4 back-to-back WMMAs (fewer dscnt wait points).
// ---------------------------------------------------------------------------

typedef _Float16 v16h __attribute__((ext_vector_type(16)));
typedef _Float16 v8h  __attribute__((ext_vector_type(8)));
typedef _Float16 v4h  __attribute__((ext_vector_type(4)));
typedef float    v8f  __attribute__((ext_vector_type(8)));

constexpr int M_DIM = 512;     // 4 * 128
constexpr int N_DIM = 11008;
constexpr int K_DIM = 4096;

constexpr int BM = 128;        // block tile M
constexpr int BN = 128;        // block tile N
constexpr int BK = 32;         // K step = WMMA K
constexpr int NT = K_DIM / BK; // 128 K-steps
constexpr int LDS_STRIDE = BK + 8;   // 40 halves (80 B) to spread banks

__global__ __launch_bounds__(256)
void QuantLinear_wmma_kernel(const float* __restrict__ A,
                             const int*   __restrict__ Q,
                             const float* __restrict__ scale,
                             const float* __restrict__ bias,
                             float*       __restrict__ out)
{
    __shared__ _Float16 As[2][BM * LDS_STRIDE];   // [buf][row m][k] f16
    __shared__ _Float16 Bs[2][BN * LDS_STRIDE];   // [buf][row o][k] f16 (= q-128)

    const int tid  = threadIdx.x;
    const int lane = tid & 31;
    const int wave = tid >> 5;        // 8 waves
    const int wm   = wave >> 2;       // 2 wave-rows of 64 M each
    const int wn   = wave & 3;        // 4 wave-cols of 32 N each
    const int l16  = lane & 15;
    const int hi   = lane >> 4;       // half-wave select

    const int mBase = blockIdx.y * BM;
    const int nBase = blockIdx.x * BN;

    // ---- accumulators: 4 (M) x 2 (N) WMMA tiles ------------------------
    v8f acc[4][2];
#pragma unroll
    for (int mi = 0; mi < 4; ++mi)
#pragma unroll
        for (int ni = 0; ni < 2; ++ni)
            acc[mi][ni] = v8f{0.f, 0.f, 0.f, 0.f, 0.f, 0.f, 0.f, 0.f};

    // ---- register stage for the in-flight global tile ------------------
    float4 aSt[4];
    int4   qSt[4];

    // Issue global loads for tile kt into registers (128 rows x 32 cols,
    // 1024 vec4; 256 threads x 4 vec4 each).
    auto stage = [&](int kt) {
        const int k0 = kt * BK;
#pragma unroll
        for (int j = 0; j < 4; ++j) {
            const int idx = tid + 256 * j;
            const int row = idx >> 3;
            const int c4  = idx & 7;
            aSt[j] = *reinterpret_cast<const float4*>(
                A + (size_t)(mBase + row) * K_DIM + k0 + c4 * 4);
            qSt[j] = *reinterpret_cast<const int4*>(
                Q + (size_t)(nBase + row) * K_DIM + k0 + c4 * 4);
        }
    };

    // Convert staged registers to f16 and store into LDS buffer pp.
    auto commit = [&](int pp) {
        _Float16* Ad = As[pp];
        _Float16* Bd = Bs[pp];
#pragma unroll
        for (int j = 0; j < 4; ++j) {
            const int idx = tid + 256 * j;
            const int row = idx >> 3;
            const int c4  = idx & 7;
            v4h ah;
            ah[0] = (_Float16)aSt[j].x;  ah[1] = (_Float16)aSt[j].y;
            ah[2] = (_Float16)aSt[j].z;  ah[3] = (_Float16)aSt[j].w;
            *reinterpret_cast<v4h*>(&Ad[row * LDS_STRIDE + c4 * 4]) = ah;
            v4h bh;
            bh[0] = (_Float16)(qSt[j].x - 128);  bh[1] = (_Float16)(qSt[j].y - 128);
            bh[2] = (_Float16)(qSt[j].z - 128);  bh[3] = (_Float16)(qSt[j].w - 128);
            *reinterpret_cast<v4h*>(&Bd[row * LDS_STRIDE + c4 * 4]) = bh;
        }
    };

    // Fragment loads + 8 WMMAs from LDS buffer pp.
    // A 16x32 f16 ISA layout: lane<16 row=l16 K0..7 & 16..23; lane>=16 K8..15 & 24..31.
    // B 32x16 f16 ISA layout: lane<16 col=l16 K0..7 & 16..23; lane>=16 K8..15 & 24..31.
    auto compute = [&](int pp) {
        const _Float16* Asb = As[pp];
        const _Float16* Bsb = Bs[pp];

        v16h afr[4];                       // resident across both B groups
#pragma unroll
        for (int mi = 0; mi < 4; ++mi) {
            const _Float16* p =
                &Asb[(wm * 64 + mi * 16 + l16) * LDS_STRIDE + hi * 8];
            v8h lo = *reinterpret_cast<const v8h*>(p);
            v8h hh = *reinterpret_cast<const v8h*>(p + 16);
            afr[mi] = __builtin_shufflevector(lo, hh,
                0, 1, 2, 3, 4, 5, 6, 7, 8, 9, 10, 11, 12, 13, 14, 15);
        }
#pragma unroll
        for (int ni = 0; ni < 2; ++ni) {
            const _Float16* p =
                &Bsb[(wn * 32 + ni * 16 + l16) * LDS_STRIDE + hi * 8];
            v8h lo = *reinterpret_cast<const v8h*>(p);
            v8h hh = *reinterpret_cast<const v8h*>(p + 16);
            v16h bfr = __builtin_shufflevector(lo, hh,
                0, 1, 2, 3, 4, 5, 6, 7, 8, 9, 10, 11, 12, 13, 14, 15);
#pragma unroll
            for (int mi = 0; mi < 4; ++mi)     // 4 back-to-back WMMAs per B frag
                acc[mi][ni] = __builtin_amdgcn_wmma_f32_16x16x32_f16(
                    /*neg_a=*/false, afr[mi],
                    /*neg_b=*/false, bfr,
                    /*c_mod=*/(short)0, acc[mi][ni],
                    /*reuse_a=*/false, /*reuse_b=*/false);
        }
    };

    // ---- pipelined main loop -------------------------------------------
    stage(0);
    commit(0);
    __syncthreads();

    int p = 0;
    for (int kt = 0; kt < NT - 1; ++kt) {
        stage(kt + 1);      // global loads in flight during compute
        compute(p);         // ds reads + 8 WMMAs on current buffer
        commit(1 - p);      // convert + LDS store of next tile
        __syncthreads();    // one barrier per K-step
        p ^= 1;
    }
    compute(p);             // last tile

    // ---- epilogue: out = scale[o]*acc + bias[o] ------------------------
    // C/D layout: VGPR r -> M = r + 8*hi ; lane%16 -> N.
#pragma unroll
    for (int ni = 0; ni < 2; ++ni) {
        const int col = nBase + wn * 32 + ni * 16 + l16;
        const float s  = scale[col];
        const float bv = bias[col];
#pragma unroll
        for (int mi = 0; mi < 4; ++mi) {
            const int rowBase = mBase + wm * 64 + mi * 16 + hi * 8;
#pragma unroll
            for (int r = 0; r < 8; ++r) {
                out[(size_t)(rowBase + r) * N_DIM + col] =
                    s * acc[mi][ni][r] + bv;
            }
        }
    }
}

extern "C" void kernel_launch(void* const* d_in, const int* in_sizes, int n_in,
                              void* d_out, int out_size, void* d_ws, size_t ws_size,
                              hipStream_t stream) {
    (void)in_sizes; (void)n_in; (void)out_size; (void)d_ws; (void)ws_size;

    const float* input   = (const float*)d_in[0];   // [4,128,4096] f32
    const int*   qweight = (const int*)  d_in[1];   // [11008,4096] i32
    const float* scale   = (const float*)d_in[2];   // [11008,1] f32
    const float* bias    = (const float*)d_in[3];   // [11008] f32
    float*       out     = (float*)d_out;           // [4,128,11008] f32

    dim3 grid(N_DIM / BN, M_DIM / BM);   // (86, 4)
    dim3 block(256);                     // 8 wave32 waves
    QuantLinear_wmma_kernel<<<grid, block, 0, stream>>>(
        input, qweight, scale, bias, out);
}
